// GAT_1056561955306
// MI455X (gfx1250) — matrix operand order repaired and checked by reading, not compiled, not run on hardware
//
#include <hip/hip_runtime.h>
#include <math.h>

typedef __attribute__((ext_vector_type(16))) _Float16 v16h;
typedef __attribute__((ext_vector_type(8)))  _Float16 v8h;
typedef __attribute__((ext_vector_type(8)))  float    v8f;

#define N_NODES  50000
#define N_EDGES  800000
#define E_TOT    (N_EDGES + N_NODES)   // edges + self loops
#define IN_CH    128
#define NCOL     256                    // [W1l | W1r] fused columns
#define N_GRAPHS 64
#define CEILDIV(a,b) (((a)+(b)-1)/(b))

union AV { v16h v; v8h h[2]; };

// ---------- monotone float<->int mapping for atomicMax on floats ----------
__device__ __forceinline__ int fmax_map(float f) {
    int i = __float_as_int(f);
    return i >= 0 ? i : (i ^ 0x7FFFFFFF);
}
__device__ __forceinline__ float fmax_unmap(int i) {
    return __int_as_float(i >= 0 ? i : (i ^ 0x7FFFFFFF));
}
__device__ __forceinline__ float lrelu(float v) { return v > 0.f ? v : 0.2f * v; }

// -------------------------------- fills --------------------------------
__global__ void fill_f32(float* p, float v, long long n) {
    long long i = (long long)blockIdx.x * blockDim.x + threadIdx.x;
    if (i < n) p[i] = v;
}
__global__ void fill_i32(int* p, int v, long long n) {
    long long i = (long long)blockIdx.x * blockDim.x + threadIdx.x;
    if (i < n) p[i] = v;
}

// --------------------------- f32 -> f16 staging ---------------------------
__global__ void cvt_x_f16(const float* __restrict__ x, _Float16* __restrict__ xf, long long n) {
    long long i = (long long)blockIdx.x * blockDim.x + threadIdx.x;
    if (i < n) xf[i] = (_Float16)x[i];
}
// Wt[n][k] = (n<128 ? W1l[k][n] : W1r[k][n-128]), transposed so B columns are contiguous
__global__ void build_wt_f16(const float* __restrict__ W1l, const float* __restrict__ W1r,
                             _Float16* __restrict__ wt) {
    int i = blockIdx.x * blockDim.x + threadIdx.x;
    if (i >= NCOL * IN_CH) return;
    int n = i >> 7, k = i & 127;
    float v = (n < 128) ? W1l[k * 128 + n] : W1r[k * 128 + (n - 128)];
    wt[n * IN_CH + k] = (_Float16)v;
}

// ------------------- GEMM1: [50000,128]x[128,256] via WMMA -------------------
// one wave per 16x16 output tile; K=128 = 4 x v_wmma_f32_16x16x32_f16
__global__ void __launch_bounds__(256)
gemm1_wmma(const _Float16* __restrict__ X, const _Float16* __restrict__ Wt,
           float* __restrict__ C) {
    int wave = (blockIdx.x * blockDim.x + threadIdx.x) >> 5;   // 0..49999
    int lane = threadIdx.x & 31;
    int mt = wave >> 4;          // 0..3124
    int nt = wave & 15;          // 0..15
    int l15 = lane & 15;
    int hi  = lane >> 4;         // lane half

    const _Float16* xr = X  + (size_t)(mt * 16 + l15) * IN_CH;
    const _Float16* wr = Wt + (size_t)(nt * 16 + l15) * IN_CH;
    int kbA = hi << 3;           // A swizzle: 0 or 8
    int kbB = hi << 3;           // B layout : 0 or 8

    v8f acc = {};
#pragma unroll
    for (int s = 0; s < 4; ++s) {
        int k0 = s * 32;
        AV a, b;
        a.h[0] = *(const v8h*)(xr + k0 + kbA);
        a.h[1] = *(const v8h*)(xr + k0 + 16 + kbA);
        b.h[0] = *(const v8h*)(wr + k0 + kbB);
        b.h[1] = *(const v8h*)(wr + k0 + 16 + kbB);
        acc = __builtin_amdgcn_wmma_f32_16x16x32_f16(false, a.v, false, b.v,
                                                     (short)0, acc, false, false);
    }
    // D layout: VGPR r -> row r (lanes 0-15) / r+8 (lanes 16-31), col = lane&15
    float* crow = C + (size_t)(mt * 16 + hi * 8) * NCOL + nt * 16 + l15;
#pragma unroll
    for (int r = 0; r < 8; ++r) crow[(size_t)r * NCOL] = acc[r];
}

// ---------------- layer-1 edge pass A: alpha + segment max ----------------
// one wave per edge; lanes 0-15 head0, 16-31 head1; 4 channels/lane
__global__ void __launch_bounds__(256)
edge1_alpha(const int* __restrict__ ei, const float* __restrict__ XLR,
            const float* __restrict__ att1, float* __restrict__ alpha1,
            int* __restrict__ amax1) {
    int e = (blockIdx.x * blockDim.x + threadIdx.x) >> 5;
    if (e >= E_TOT) return;
    int lane = threadIdx.x & 31;
    int src = (e < N_EDGES) ? ei[e] : (e - N_EDGES);
    int dst = (e < N_EDGES) ? ei[N_EDGES + e] : (e - N_EDGES);
    const float* xl = XLR + (size_t)src * NCOL;         // cols [0,128) = x@W1l
    const float* xr = XLR + (size_t)dst * NCOL + 128;   // cols [128,256) = x@W1r
    int ch0 = lane * 4;                                  // flat channel h*64+c
    float p = 0.f;
#pragma unroll
    for (int j = 0; j < 4; ++j) {
        int ch = ch0 + j;
        p += lrelu(xl[ch] + xr[ch]) * att1[ch];          // att1 flat [2*64]
    }
    p += __shfl_xor(p, 1, 32); p += __shfl_xor(p, 2, 32);
    p += __shfl_xor(p, 4, 32); p += __shfl_xor(p, 8, 32);
    if ((lane & 15) == 0) {
        int h = lane >> 4;
        alpha1[(size_t)e * 2 + h] = p;
        atomicMax(&amax1[(size_t)dst * 2 + h], fmax_map(p));
    }
}

// ------------- layer-1 edge pass B: exp(alpha - max) + denom -------------
__global__ void __launch_bounds__(256)
edge1_exp(const int* __restrict__ ei, float* __restrict__ alpha1,
          const int* __restrict__ amax1, float* __restrict__ denom1) {
    long long idx = (long long)blockIdx.x * blockDim.x + threadIdx.x;
    if (idx >= (long long)E_TOT * 2) return;
    int e = (int)(idx >> 1), h = (int)(idx & 1);
    int dst = (e < N_EDGES) ? ei[N_EDGES + e] : (e - N_EDGES);
    float ea = expf(alpha1[idx] - fmax_unmap(amax1[(size_t)dst * 2 + h]));
    alpha1[idx] = ea;
    atomicAdd(&denom1[(size_t)dst * 2 + h], ea);
}

// ---------------- layer-1 edge pass C: weighted scatter-add ----------------
__global__ void __launch_bounds__(256)
edge1_agg(const int* __restrict__ ei, const float* __restrict__ XLR,
          const float* __restrict__ alpha1, const float* __restrict__ denom1,
          float* __restrict__ h1) {
    int e = (blockIdx.x * blockDim.x + threadIdx.x) >> 5;
    if (e >= E_TOT) return;
    int lane = threadIdx.x & 31;
    int src = (e < N_EDGES) ? ei[e] : (e - N_EDGES);
    int dst = (e < N_EDGES) ? ei[N_EDGES + e] : (e - N_EDGES);
    int h = lane >> 4;
    float w = alpha1[(size_t)e * 2 + h] / (denom1[(size_t)dst * 2 + h] + 1e-16f);
    const float* xl = XLR + (size_t)src * NCOL;
    float* od = h1 + (size_t)dst * IN_CH;
    int ch0 = lane * 4;
#pragma unroll
    for (int j = 0; j < 4; ++j) {
        int ch = ch0 + j;
        atomicAdd(&od[ch], xl[ch] * w);
    }
}

// --------------------------- bias + ELU (in place) ---------------------------
__global__ void bias_elu(float* __restrict__ h1, const float* __restrict__ b1, long long n) {
    long long i = (long long)blockIdx.x * blockDim.x + threadIdx.x;
    if (i >= n) return;
    float v = h1[i] + b1[i & 127];
    h1[i] = v > 0.f ? v : (expf(v) - 1.f);
}

// ------------- GEMM2: [50000,128]x[128,3] x2 (wave per node) -------------
__global__ void __launch_bounds__(256)
gemm2(const float* __restrict__ h1, const float* __restrict__ W2l,
      const float* __restrict__ W2r, float* __restrict__ xl2, float* __restrict__ xr2) {
    int n = (blockIdx.x * blockDim.x + threadIdx.x) >> 5;
    if (n >= N_NODES) return;
    int lane = threadIdx.x & 31;
    const float* hr = h1 + (size_t)n * IN_CH;
    float pl0 = 0, pl1 = 0, pl2 = 0, pr0 = 0, pr1 = 0, pr2 = 0;
#pragma unroll
    for (int j = 0; j < 4; ++j) {
        int k = lane * 4 + j;
        float hv = hr[k];
        pl0 += hv * W2l[k * 3 + 0]; pl1 += hv * W2l[k * 3 + 1]; pl2 += hv * W2l[k * 3 + 2];
        pr0 += hv * W2r[k * 3 + 0]; pr1 += hv * W2r[k * 3 + 1]; pr2 += hv * W2r[k * 3 + 2];
    }
#pragma unroll
    for (int o = 16; o >= 1; o >>= 1) {
        pl0 += __shfl_xor(pl0, o, 32); pl1 += __shfl_xor(pl1, o, 32); pl2 += __shfl_xor(pl2, o, 32);
        pr0 += __shfl_xor(pr0, o, 32); pr1 += __shfl_xor(pr1, o, 32); pr2 += __shfl_xor(pr2, o, 32);
    }
    if (lane == 0) {
        xl2[n * 3 + 0] = pl0; xl2[n * 3 + 1] = pl1; xl2[n * 3 + 2] = pl2;
        xr2[n * 3 + 0] = pr0; xr2[n * 3 + 1] = pr1; xr2[n * 3 + 2] = pr2;
    }
}

// --------------------------- layer-2 edge passes ---------------------------
__global__ void __launch_bounds__(256)
edge2_alpha(const int* __restrict__ ei, const float* __restrict__ xl2,
            const float* __restrict__ xr2, const float* __restrict__ att2,
            float* __restrict__ alpha2, int* __restrict__ amax2) {
    int e = blockIdx.x * blockDim.x + threadIdx.x;
    if (e >= E_TOT) return;
    int src = (e < N_EDGES) ? ei[e] : (e - N_EDGES);
    int dst = (e < N_EDGES) ? ei[N_EDGES + e] : (e - N_EDGES);
    float a = 0.f;
#pragma unroll
    for (int c = 0; c < 3; ++c)
        a += lrelu(xl2[src * 3 + c] + xr2[dst * 3 + c]) * att2[c];
    alpha2[e] = a;
    atomicMax(&amax2[dst], fmax_map(a));
}

__global__ void __launch_bounds__(256)
edge2_exp(const int* __restrict__ ei, float* __restrict__ alpha2,
          const int* __restrict__ amax2, float* __restrict__ denom2) {
    int e = blockIdx.x * blockDim.x + threadIdx.x;
    if (e >= E_TOT) return;
    int dst = (e < N_EDGES) ? ei[N_EDGES + e] : (e - N_EDGES);
    float ea = expf(alpha2[e] - fmax_unmap(amax2[dst]));
    alpha2[e] = ea;
    atomicAdd(&denom2[dst], ea);
}

__global__ void __launch_bounds__(256)
edge2_agg(const int* __restrict__ ei, const float* __restrict__ xl2,
          const float* __restrict__ alpha2, const float* __restrict__ denom2,
          float* __restrict__ h2) {
    int e = blockIdx.x * blockDim.x + threadIdx.x;
    if (e >= E_TOT) return;
    int src = (e < N_EDGES) ? ei[e] : (e - N_EDGES);
    int dst = (e < N_EDGES) ? ei[N_EDGES + e] : (e - N_EDGES);
    float w = alpha2[e] / (denom2[dst] + 1e-16f);
#pragma unroll
    for (int c = 0; c < 3; ++c)
        atomicAdd(&h2[dst * 3 + c], xl2[src * 3 + c] * w);
}

// ---------------------------- global mean pool ----------------------------
__global__ void __launch_bounds__(256)
pool(const float* __restrict__ h2, const float* __restrict__ b2,
     const int* __restrict__ batch, float* __restrict__ gsum, float* __restrict__ gcnt) {
    int n = blockIdx.x * blockDim.x + threadIdx.x;
    if (n >= N_NODES) return;
    int b = batch[n];
#pragma unroll
    for (int c = 0; c < 3; ++c)
        atomicAdd(&gsum[b * 3 + c], h2[n * 3 + c] + b2[c]);
    atomicAdd(&gcnt[b], 1.f);
}

// ------------------------------- readout MLP -------------------------------
__global__ void mlp(const float* __restrict__ gsum, const float* __restrict__ gcnt,
                    const float* __restrict__ Wr1, const float* __restrict__ br1,
                    const float* __restrict__ Wr2, const float* __restrict__ br2,
                    float* __restrict__ out) {
    int g = threadIdx.x;
    if (g >= N_GRAPHS) return;
    float cnt = gcnt[g]; cnt = cnt > 1.f ? cnt : 1.f;
    float x0 = gsum[g * 3 + 0] / cnt, x1 = gsum[g * 3 + 1] / cnt, x2 = gsum[g * 3 + 2] / cnt;
    float o0 = br2[0], o1 = br2[1], o2 = br2[2];
#pragma unroll 4
    for (int j = 0; j < 64; ++j) {
        float hj = x0 * Wr1[j] + x1 * Wr1[64 + j] + x2 * Wr1[128 + j] + br1[j];
        hj = hj > 0.f ? hj : 0.f;
        o0 += hj * Wr2[j * 3 + 0]; o1 += hj * Wr2[j * 3 + 1]; o2 += hj * Wr2[j * 3 + 2];
    }
    out[g * 3 + 0] = o0; out[g * 3 + 1] = o1; out[g * 3 + 2] = o2;
}

// =============================== launcher ===============================
extern "C" void kernel_launch(void* const* d_in, const int* in_sizes, int n_in,
                              void* d_out, int out_size, void* d_ws, size_t ws_size,
                              hipStream_t stream) {
    const float* x     = (const float*)d_in[0];
    const int*   ei    = (const int*)  d_in[1];
    const int*   batch = (const int*)  d_in[2];
    const float* W1l   = (const float*)d_in[3];
    const float* W1r   = (const float*)d_in[4];
    const float* att1  = (const float*)d_in[5];
    const float* b1    = (const float*)d_in[6];
    const float* W2l   = (const float*)d_in[7];
    const float* W2r   = (const float*)d_in[8];
    const float* att2  = (const float*)d_in[9];
    const float* b2    = (const float*)d_in[10];
    const float* Wr1   = (const float*)d_in[11];
    const float* br1   = (const float*)d_in[12];
    const float* Wr2   = (const float*)d_in[13];
    const float* br2   = (const float*)d_in[14];
    float* out = (float*)d_out;

    char* ws = (char*)d_ws;
    size_t off = 0;
    auto sub = [&](size_t bytes) -> char* {
        char* p = ws + off; off += (bytes + 255) & ~(size_t)255; return p;
    };
    float* XLR    = (float*)sub((size_t)N_NODES * NCOL * 4);    // 51.2 MB
    float* h1     = (float*)sub((size_t)N_NODES * IN_CH * 4);   // 25.6 MB
    float* alpha1 = (float*)sub((size_t)E_TOT * 2 * 4);         //  6.8 MB
    int*   amax1  = (int*)  sub((size_t)N_NODES * 2 * 4);
    float* denom1 = (float*)sub((size_t)N_NODES * 2 * 4);
    // staging region (f16 X + Wt), recycled for all layer-2 buffers after GEMM1
    size_t regA = off;
    _Float16* xf16 = (_Float16*)sub((size_t)N_NODES * IN_CH * 2);  // 12.8 MB
    _Float16* wt   = (_Float16*)sub((size_t)NCOL * IN_CH * 2);
    size_t off2 = regA;
    auto sub2 = [&](size_t bytes) -> char* {
        char* p = ws + off2; off2 += (bytes + 255) & ~(size_t)255; return p;
    };
    float* alpha2 = (float*)sub2((size_t)E_TOT * 4);            // 3.4 MB (fits region)
    float* xl2    = (float*)sub2((size_t)N_NODES * 3 * 4);
    float* xr2    = (float*)sub2((size_t)N_NODES * 3 * 4);
    int*   amax2  = (int*)  sub2((size_t)N_NODES * 4);
    float* denom2 = (float*)sub2((size_t)N_NODES * 4);
    float* h2     = (float*)sub2((size_t)N_NODES * 3 * 4);
    float* gsum   = (float*)sub2((size_t)N_GRAPHS * 3 * 4);
    float* gcnt   = (float*)sub2((size_t)N_GRAPHS * 4);

    const int T = 256;
    // layer-1 accumulators
    fill_f32<<<CEILDIV((long long)N_NODES * IN_CH, T), T, 0, stream>>>(h1, 0.f, (long long)N_NODES * IN_CH);
    fill_i32<<<CEILDIV(N_NODES * 2, T), T, 0, stream>>>(amax1, INT_MIN, N_NODES * 2);
    fill_f32<<<CEILDIV(N_NODES * 2, T), T, 0, stream>>>(denom1, 0.f, N_NODES * 2);
    // f16 staging + WMMA GEMM1
    cvt_x_f16<<<CEILDIV((long long)N_NODES * IN_CH, T), T, 0, stream>>>(x, xf16, (long long)N_NODES * IN_CH);
    build_wt_f16<<<CEILDIV(NCOL * IN_CH, T), T, 0, stream>>>(W1l, W1r, wt);
    gemm1_wmma<<<(N_NODES * 16) / (T / 32) / 16, T, 0, stream>>>(xf16, wt, XLR); // 6250 blocks
    // layer-2 accumulators (staging region now dead)
    fill_i32<<<CEILDIV(N_NODES, T), T, 0, stream>>>(amax2, INT_MIN, N_NODES);
    fill_f32<<<CEILDIV(N_NODES, T), T, 0, stream>>>(denom2, 0.f, N_NODES);
    fill_f32<<<CEILDIV(N_NODES * 3, T), T, 0, stream>>>(h2, 0.f, N_NODES * 3);
    fill_f32<<<1, T, 0, stream>>>(gsum, 0.f, N_GRAPHS * 3);
    fill_f32<<<1, T, 0, stream>>>(gcnt, 0.f, N_GRAPHS);
    // layer-1 attention
    edge1_alpha<<<CEILDIV(E_TOT, 8), T, 0, stream>>>(ei, XLR, att1, alpha1, amax1);
    edge1_exp<<<CEILDIV((long long)E_TOT * 2, T), T, 0, stream>>>(ei, alpha1, amax1, denom1);
    edge1_agg<<<CEILDIV(E_TOT, 8), T, 0, stream>>>(ei, XLR, alpha1, denom1, h1);
    bias_elu<<<CEILDIV((long long)N_NODES * IN_CH, T), T, 0, stream>>>(h1, b1, (long long)N_NODES * IN_CH);
    // layer-2
    gemm2<<<CEILDIV(N_NODES, 8), T, 0, stream>>>(h1, W2l, W2r, xl2, xr2);
    edge2_alpha<<<CEILDIV(E_TOT, T), T, 0, stream>>>(ei, xl2, xr2, att2, alpha2, amax2);
    edge2_exp<<<CEILDIV(E_TOT, T), T, 0, stream>>>(ei, alpha2, amax2, denom2);
    edge2_agg<<<CEILDIV(E_TOT, T), T, 0, stream>>>(ei, xl2, alpha2, denom2, h2);
    // readout
    pool<<<CEILDIV(N_NODES, T), T, 0, stream>>>(h2, b2, batch, gsum, gcnt);
    mlp<<<1, 64, 0, stream>>>(gsum, gcnt, Wr1, br1, Wr2, br2, out);
}
